// GNODec_69140383531670
// MI455X (gfx1250) — compile-verified, matrix operand order checked by reading, not run on hardware
//
#include <hip/hip_runtime.h>
#include <hip/hip_bf16.h>

// ---------------------------------------------------------------------------
// NNConv (PyG-style) on MI455X / gfx1250.
//   edge MLP (70->20->20->20->192) : WMMA f16 (v_wmma_f32_16x16x32_f16)
//     - edge_attr tile staged to LDS with coalesced loads (rows contiguous)
//     - weights in LDS transposed [N][K] -> B frags = 2x ds_load_b128
//     - activations in LDS [M][K]        -> A frags = 2x ds_load_b128
//   per-edge einsum + scatter-mean : VALU + global_load_b128 + atomic_add_f32
//   finalize (mean + root + bias)  : VALU
// ---------------------------------------------------------------------------

typedef __attribute__((ext_vector_type(16))) _Float16 v16h;
typedef __attribute__((ext_vector_type(8)))  _Float16 v8h;
typedef __attribute__((ext_vector_type(8)))  float    v8f;

#define LATENT   64
#define EDGE_IN  70
#define K1PAD    96       // EDGE_IN padded to 3x32
#define HID      20
#define OUTD     192      // LATENT * OUT_CH
#define OUT_CH   3
#define WAVES    4        // waves per block (128 threads)

// ---------------- WMMA helpers --------------------------------------------

__device__ inline v8f wmma16(v16h a, v16h b, v8f c) {
    return __builtin_amdgcn_wmma_f32_16x16x32_f16(
        false, a, false, b, (short)0, c, false, false);
}

__device__ inline v16h cat8(v8h lo, v8h hi) {
    return __builtin_shufflevector(lo, hi,
        0, 1, 2, 3, 4, 5, 6, 7, 8, 9, 10, 11, 12, 13, 14, 15);
}

// A fragment (16x32 MxK) from LDS activation tile H[16][ld] (f16 row-major,
// ld multiple of 8). Lane L<16: M=L, K in {k0..k0+7} U {k0+16..k0+23};
// lanes 16..31 add 8 to the K base. Two 16-byte contiguous runs per lane.
__device__ inline v16h load_a(const _Float16* H, int ld, int k0) {
    int lane = threadIdx.x & 31;
    int m  = lane & 15;
    int kb = k0 + ((lane & 16) ? 8 : 0);
    v8h lo = *(const v8h*)(H + m * ld + kb);
    v8h hi = *(const v8h*)(H + m * ld + kb + 16);
    return cat8(lo, hi);
}

// B fragment (32x16 KxN) from LDS transposed weights WT[N][ldk] (f16).
__device__ inline v16h make_b(const _Float16* WT, int ldk, int k0, int n0) {
    int lane = threadIdx.x & 31;
    int n  = n0 + (lane & 15);
    int kb = k0 + ((lane & 16) ? 8 : 0);
    v8h lo = *(const v8h*)(WT + n * ldk + kb);
    v8h hi = *(const v8h*)(WT + n * ldk + kb + 16);
    return cat8(lo, hi);
}

// Store C/D fragment (16x16 f32) into LDS tile H[16][ld] as f16 with bias
// (+ optional ReLU). D layout: lane<16 -> n=lane, M=r; lane>=16 -> M=r+8.
__device__ inline void store_c(_Float16* H, int ld, v8f c, int n0,
                               const float* bias, bool relu) {
    int lane = threadIdx.x & 31;
    int n  = n0 + (lane & 15);
    int mb = (lane & 16) ? 8 : 0;
    float bv = bias[n];
#pragma unroll
    for (int r = 0; r < 8; ++r) {
        float v = c[r] + bv;
        if (relu) v = fmaxf(v, 0.0f);
        H[(mb + r) * ld + n] = (_Float16)v;
    }
}

// ---------------- Kernels --------------------------------------------------

__global__ void zero_f32(float* p, int n) {
    int i = blockIdx.x * blockDim.x + threadIdx.x;
    if (i < n) p[i] = 0.0f;
}

__global__ __launch_bounds__(WAVES * 32)
void nnconv_edge_kernel(const float* __restrict__ x,
                        const int*   __restrict__ edge_index,
                        const float* __restrict__ edge_attr,
                        const float* __restrict__ W1, const float* __restrict__ b1,
                        const float* __restrict__ W2, const float* __restrict__ b2,
                        const float* __restrict__ W3, const float* __restrict__ b3,
                        const float* __restrict__ W4, const float* __restrict__ b4,
                        float* __restrict__ summed, float* __restrict__ count,
                        int E) {
    // Transposed f16 weights [N][K], zero-padded; K-contiguous for b128 loads.
    __shared__ alignas(16) _Float16 sW1T[32  * K1PAD];  // [n][k] 32x96
    __shared__ alignas(16) _Float16 sW2T[32  * 32];
    __shared__ alignas(16) _Float16 sW3T[32  * 32];
    __shared__ alignas(16) _Float16 sW4T[OUTD * 32];    // [n][k] 192x32
    __shared__ float sB1[32], sB2[32], sB3[32];
    __shared__ float sB4[OUTD];
    // Per-wave tiles: sA doubles as layer-1 input [16][96] and then as the
    // inter-layer activation buffer H [16][32] (first third of it).
    __shared__ alignas(16) _Float16 sA [WAVES][16 * K1PAD];
    __shared__ alignas(16) _Float16 sO4[WAVES][16 * OUTD];

    const int tid = threadIdx.x;

    // ---- cooperative transposed weight load (fp32 -> f16, zero padded) ----
    for (int i = tid; i < 32 * K1PAD; i += blockDim.x) {
        int n = i / K1PAD, k = i % K1PAD;
        sW1T[i] = (_Float16)((k < EDGE_IN && n < HID) ? W1[k * HID + n] : 0.0f);
    }
    for (int i = tid; i < 32 * 32; i += blockDim.x) {
        int n = i >> 5, k = i & 31;
        bool ok = (k < HID && n < HID);
        sW2T[i] = (_Float16)(ok ? W2[k * HID + n] : 0.0f);
        sW3T[i] = (_Float16)(ok ? W3[k * HID + n] : 0.0f);
    }
    for (int i = tid; i < OUTD * 32; i += blockDim.x) {
        int n = i >> 5, k = i & 31;
        sW4T[i] = (_Float16)((k < HID) ? W4[k * OUTD + n] : 0.0f);
    }
    for (int i = tid; i < 32; i += blockDim.x) {
        sB1[i] = (i < HID) ? b1[i] : 0.0f;
        sB2[i] = (i < HID) ? b2[i] : 0.0f;
        sB3[i] = (i < HID) ? b3[i] : 0.0f;
    }
    for (int i = tid; i < OUTD; i += blockDim.x) sB4[i] = b4[i];
    __syncthreads();

    const int lane = tid & 31;
    const int wave = tid >> 5;
    const int m    = lane & 15;       // edge-in-tile (matrix row M)
    const int half = lane >> 4;       // latent half for the einsum

    const long tile = (long)blockIdx.x * WAVES + wave;
    const long e0   = tile * 16;
    if (e0 >= E) return;              // wave-uniform: EXEC all-1s below

    _Float16* A1 = sA [wave];         // [16][96]
    _Float16* O4 = sO4[wave];         // [16][192]

    // ---- stage edge_attr tile: 16 consecutive rows = 1120 contiguous f32 ----
    {
        const long gbase = e0 * EDGE_IN;
        const long gmax  = (long)E * EDGE_IN;
#pragma unroll 5
        for (int i = lane; i < 16 * EDGE_IN; i += 32) {
            long g = gbase + i;
            float v = (g < gmax) ? edge_attr[g] : 0.0f;
            A1[(i / EDGE_IN) * K1PAD + (i % EDGE_IN)] = (_Float16)v;
        }
        // zero the K padding (cols 70..95)
#pragma unroll
        for (int i = lane; i < 16 * (K1PAD - EDGE_IN); i += 32) {
            A1[(i / 26) * K1PAD + EDGE_IN + (i % 26)] = (_Float16)0.0f;
        }
    }
    asm volatile("s_wait_dscnt 0" ::: "memory");

    // ---- layer 1: [16 x 96] @ [96 x 32] ----
    _Float16* H = A1;                 // reuse staging tile as H [16][32]
    {
        v8f c0 = {}, c1 = {};
#pragma unroll
        for (int ks = 0; ks < 3; ++ks) {
            v16h a = load_a(A1, K1PAD, ks * 32);
            c0 = wmma16(a, make_b(sW1T, K1PAD, ks * 32, 0),  c0);
            c1 = wmma16(a, make_b(sW1T, K1PAD, ks * 32, 16), c1);
        }
        store_c(H, 32, c0, 0,  sB1, true);
        store_c(H, 32, c1, 16, sB1, true);
    }
    asm volatile("s_wait_dscnt 0" ::: "memory");

    // ---- layer 2 ----
    {
        v16h a = load_a(H, 32, 0);
        v8f c0 = {}, c1 = {};
        c0 = wmma16(a, make_b(sW2T, 32, 0, 0),  c0);
        c1 = wmma16(a, make_b(sW2T, 32, 0, 16), c1);
        asm volatile("s_wait_dscnt 0" ::: "memory");
        store_c(H, 32, c0, 0,  sB2, true);
        store_c(H, 32, c1, 16, sB2, true);
    }
    asm volatile("s_wait_dscnt 0" ::: "memory");

    // ---- layer 3 ----
    {
        v16h a = load_a(H, 32, 0);
        v8f c0 = {}, c1 = {};
        c0 = wmma16(a, make_b(sW3T, 32, 0, 0),  c0);
        c1 = wmma16(a, make_b(sW3T, 32, 0, 16), c1);
        asm volatile("s_wait_dscnt 0" ::: "memory");
        store_c(H, 32, c0, 0,  sB3, true);
        store_c(H, 32, c1, 16, sB3, true);
    }
    asm volatile("s_wait_dscnt 0" ::: "memory");

    // ---- layer 4: [16 x 32] @ [32 x 192] -> per-edge weight matrices ----
    {
        v16h a = load_a(H, 32, 0);
#pragma unroll
        for (int t = 0; t < OUTD / 16; ++t) {
            v8f c = {};
            c = wmma16(a, make_b(sW4T, 32, 0, t * 16), c);
            store_c(O4, OUTD, c, t * 16, sB4, false);
        }
    }
    asm volatile("s_wait_dscnt 0" ::: "memory");

    // ---- per-edge einsum msg[e,c] = sum_i x[src[e],i] * O4[e, i*3+c] ----
    // 2 lanes per edge; each lane reduces 32 contiguous latent elements:
    //   x     : 8x global_load_b128
    //   O4 row: 12x ds_load_b128 (96 contiguous halves per lane)
    const long e = e0 + m;
    if (e < E) {
        const int src = edge_index[e];
        const int dst = edge_index[(size_t)E + e];

        float xv[32];
        const float4* xr = (const float4*)(x + (size_t)src * LATENT + half * 32);
#pragma unroll
        for (int q = 0; q < 8; ++q) {
            float4 t = xr[q];
            xv[q * 4 + 0] = t.x; xv[q * 4 + 1] = t.y;
            xv[q * 4 + 2] = t.z; xv[q * 4 + 3] = t.w;
        }

        const v8h* wp = (const v8h*)(O4 + m * OUTD + half * 32 * OUT_CH);
        float p0 = 0.0f, p1 = 0.0f, p2 = 0.0f;
#pragma unroll
        for (int q = 0; q < 12; ++q) {
            v8h w8 = wp[q];
#pragma unroll
            for (int j = 0; j < 8; ++j) {
                int t = q * 8 + j;          // flat index into 32x3 block
                int i = t / 3, c = t % 3;   // compile-time after unroll
                float prod = xv[i] * (float)w8[j];
                if (c == 0) p0 += prod;
                else if (c == 1) p1 += prod;
                else p2 += prod;
            }
        }
        p0 += __shfl_xor(p0, 16);
        p1 += __shfl_xor(p1, 16);
        p2 += __shfl_xor(p2, 16);
        if (half == 0) {
            atomicAdd(&summed[(size_t)dst * 3 + 0], p0);
            atomicAdd(&summed[(size_t)dst * 3 + 1], p1);
            atomicAdd(&summed[(size_t)dst * 3 + 2], p2);
            atomicAdd(&count[dst], 1.0f);
        }
    }
}

__global__ void nnconv_finalize_kernel(const float* __restrict__ x,
                                       const float* __restrict__ root_w,
                                       const float* __restrict__ bias,
                                       const float* __restrict__ summed,
                                       const float* __restrict__ count,
                                       float* __restrict__ out, int N) {
    int n = blockIdx.x * blockDim.x + threadIdx.x;
    if (n >= N) return;
    float inv = 1.0f / fmaxf(count[n], 1.0f);
    float a0 = summed[(size_t)n * 3 + 0] * inv + bias[0];
    float a1 = summed[(size_t)n * 3 + 1] * inv + bias[1];
    float a2 = summed[(size_t)n * 3 + 2] * inv + bias[2];
    const float4* xr = (const float4*)(x + (size_t)n * LATENT);
#pragma unroll
    for (int q = 0; q < LATENT / 4; ++q) {
        float4 v = xr[q];
        const float* vp = (const float*)&v;
#pragma unroll
        for (int j = 0; j < 4; ++j) {
            int i = q * 4 + j;
            float xvv = vp[j];
            a0 += xvv * root_w[i * 3 + 0];
            a1 += xvv * root_w[i * 3 + 1];
            a2 += xvv * root_w[i * 3 + 2];
        }
    }
    out[(size_t)n * 3 + 0] = a0;
    out[(size_t)n * 3 + 1] = a1;
    out[(size_t)n * 3 + 2] = a2;
}

// ---------------- Host launch ---------------------------------------------

extern "C" void kernel_launch(void* const* d_in, const int* in_sizes, int n_in,
                              void* d_out, int out_size, void* d_ws, size_t ws_size,
                              hipStream_t stream) {
    const float* x          = (const float*)d_in[0];
    const int*   edge_index = (const int*)  d_in[1];
    const float* edge_attr  = (const float*)d_in[2];
    const float* W1 = (const float*)d_in[3];
    const float* b1 = (const float*)d_in[4];
    const float* W2 = (const float*)d_in[5];
    const float* b2 = (const float*)d_in[6];
    const float* W3 = (const float*)d_in[7];
    const float* b3 = (const float*)d_in[8];
    const float* W4 = (const float*)d_in[9];
    const float* b4 = (const float*)d_in[10];
    const float* root_w = (const float*)d_in[11];
    const float* bias   = (const float*)d_in[12];
    float* out = (float*)d_out;

    const int E = in_sizes[1] / 2;
    const int N = in_sizes[0] / LATENT;

    float* summed = (float*)d_ws;            // [N,3]
    float* count  = summed + (size_t)N * 3;  // [N]

    {   // zero accumulators (summed + count contiguous)
        int zn = N * 4;
        zero_f32<<<(zn + 255) / 256, 256, 0, stream>>>(summed, zn);
    }
    {   // edge MLP + message + scatter
        int tiles  = (E + 15) / 16;
        int blocks = (tiles + WAVES - 1) / WAVES;
        nnconv_edge_kernel<<<blocks, WAVES * 32, 0, stream>>>(
            x, edge_index, edge_attr,
            W1, b1, W2, b2, W3, b3, W4, b4,
            summed, count, E);
    }
    nnconv_finalize_kernel<<<(N + 127) / 128, 128, 0, stream>>>(
        x, root_w, bias, summed, count, out, N);
}
